// VGG_43619687858769
// MI455X (gfx1250) — matrix operand order repaired and checked by reading, not compiled, not run on hardware
//
#include <hip/hip_runtime.h>
#include <hip/hip_bf16.h>

typedef __bf16 bf16;
typedef __attribute__((ext_vector_type(16))) __bf16 v16bf;
typedef __attribute__((ext_vector_type(8)))  float  v8f;
typedef __attribute__((ext_vector_type(2)))  int    v2i;

// CDNA5 async-to-LDS path (probe via __has_builtin; clean fallback otherwise)
#if defined(__AMDGCN__) && __has_builtin(__builtin_amdgcn_global_load_async_to_lds_b64) && \
    __has_builtin(__builtin_amdgcn_s_wait_asynccnt)
#define USE_ASYNC 1
typedef __attribute__((address_space(1))) v2i gv2i;   // global b64 payload ptr
typedef __attribute__((address_space(3))) v2i lv2i;   // LDS b64 payload ptr
#else
#define USE_ASYNC 0
#endif

// Assemble a 16-element bf16 fragment (8 VGPRs) from two 16-byte LDS reads.
__device__ inline v16bf ld_frag(const bf16* lo, const bf16* hi) {
    struct P { uint4 a, b; } p;
    p.a = *(const uint4*)lo;
    p.b = *(const uint4*)hi;
    return __builtin_bit_cast(v16bf, p);
}

// ---------------------------------------------------------------------------
// Implicit-GEMM 3x3 conv (stride 1, pad 1) + ReLU, bf16 out, f32 accumulate.
// Compile-time C,H -> Ktot, HW constexpr: no guards, div/mod become shifts.
// Block tile: 64 oc x 64 px; 8 waves; each wave: 1 A-frag x 2 B-frags = 2 WMMA
// per 32-wide K chunk. Weights read fp32 (reside in 192MB L2), cvt at staging.
// ---------------------------------------------------------------------------
template<typename TIN, int C, int H>
__global__ __launch_bounds__(256)
void conv3x3_wmma_kernel(const TIN* __restrict__ in, const float* __restrict__ w,
                         bf16* __restrict__ out, int OC)
{
    constexpr int W    = H;
    constexpr int HW   = H * W;
    constexpr int Ktot = C * 9;
    constexpr bool KGUARD = (Ktot % 32) != 0;   // only the C==3 first layer

    // rows padded to 40 elems (80 B): every b128 access stays 16-B aligned
    __shared__ __align__(16) bf16 As[64 * 40];  // [oc 0..63][k 0..31]
    __shared__ __align__(16) bf16 Bs[64 * 40];  // [pixel 0..63][k 0..31]

    const int t      = threadIdx.x;
    const int lane   = t & 31;
    const int wv     = t >> 5;
    const int waveOc = wv & 3;                  // 4 oc sub-tiles of 16
    const int wavePx = wv >> 2;                 // 2 px sub-tiles of 32 (2 wmma)

    const int pBase  = blockIdx.x * 64;         // Ptot % 64 == 0 always
    const int ocBase = blockIdx.y * 64;         // OC % 64 == 0 always

    // B staging: 64 px x 32 k / 256 thr -> 8 consecutive k for one pixel
    const int pl  = t >> 2;                     // 0..63
    const int kb  = (t & 3) * 8;                // 0,8,16,24
    const int p   = pBase + pl;
    const int n   = p / HW;                     // constexpr HW -> shift
    const int rem = p & (HW - 1);
    const int y   = rem / W;
    const int x   = rem & (W - 1);

    // A staging: 64 oc x 32 k / 256 thr -> 8 consecutive k for one oc row
    const int ar  = t >> 2;
    const int ac  = (t & 3) * 8;
    const int aoc = ocBase + ar;

    // fragment pointers (loop-invariant)
    const int m = lane & 15, half = lane >> 4;
    const bf16* arow  = &As[(waveOc * 16 + m) * 40];
    const bf16* brow0 = &Bs[(wavePx * 32 + m) * 40];
    const bf16* brow1 = &Bs[(wavePx * 32 + 16 + m) * 40];

    v8f acc[2] = {{}, {}};

    for (int k0 = 0; k0 < Ktot; k0 += 32) {
        __syncthreads();
        // ---- stage A: weights fp32 -> bf16, vectorized b128 ----
        union { bf16 h[8]; uint4 q; } ua;
        if constexpr (KGUARD) {
            #pragma unroll
            for (int j = 0; j < 8; ++j) {
                int k = k0 + ac + j;
                ua.h[j] = (bf16)((k < Ktot) ? w[aoc * Ktot + k] : 0.0f);
            }
        } else {
            const float4 f0 = *(const float4*)(w + aoc * Ktot + k0 + ac);
            const float4 f1 = *(const float4*)(w + aoc * Ktot + k0 + ac + 4);
            ua.h[0] = (bf16)f0.x; ua.h[1] = (bf16)f0.y;
            ua.h[2] = (bf16)f0.z; ua.h[3] = (bf16)f0.w;
            ua.h[4] = (bf16)f1.x; ua.h[5] = (bf16)f1.y;
            ua.h[6] = (bf16)f1.z; ua.h[7] = (bf16)f1.w;
        }
        *(uint4*)&As[ar * 40 + ac] = ua.q;

        // ---- stage B: branch-free im2col gather (clamped addr + cndmask) ----
        union { bf16 h[8]; uint4 q; } ub;
        #pragma unroll
        for (int j = 0; j < 8; ++j) {
            int k  = k0 + kb + j;
            int ic = k / 9;
            int r  = k - ic * 9;
            int ky = r / 3;
            int kx = r - ky * 3;
            int yy = y + ky - 1;
            int xx = x + kx - 1;
            bool ok = ((unsigned)yy < (unsigned)H) & ((unsigned)xx < (unsigned)W);
            if (KGUARD) ok = ok && (k < Ktot);
            int icc = ok ? ic : 0;
            int yyc = ok ? yy : 0;
            int xxc = ok ? xx : 0;
            float v = (float)in[((n * C + icc) * H + yyc) * W + xxc];
            ub.h[j] = (bf16)(ok ? v : 0.0f);
        }
        *(uint4*)&Bs[pl * 40 + kb] = ub.q;

        __syncthreads();

        // ---- 2x WMMA 16x16x32 bf16, A fragment reused ----
        v16bf a  = ld_frag(arow  + half * 8,  arow  + 16 + half * 8);
        v16bf b0 = ld_frag(brow0 + half * 8, brow0 + 16 + half * 8);
        v16bf b1 = ld_frag(brow1 + half * 8, brow1 + 16 + half * 8);
        acc[0] = __builtin_amdgcn_wmma_f32_16x16x32_bf16(false, a, false, b0,
                                                         (short)0, acc[0], false, false);
        acc[1] = __builtin_amdgcn_wmma_f32_16x16x32_bf16(false, a, false, b1,
                                                         (short)0, acc[1], false, false);
    }

    // ---- epilogue: ReLU, fp32 -> bf16, NCHW store ----
    #pragma unroll
    for (int tpx = 0; tpx < 2; ++tpx) {
        const int sp   = pBase + wavePx * 32 + tpx * 16 + m;
        const int sn   = sp / HW;
        const int srem = sp & (HW - 1);
        const int sy   = srem / W;
        const int sx   = srem & (W - 1);
        #pragma unroll
        for (int r = 0; r < 8; ++r) {
            const int oc = ocBase + waveOc * 16 + ((lane < 16) ? r : r + 8);
            float v = acc[tpx][r];
            v = v > 0.0f ? v : 0.0f;
            out[((sn * OC + oc) * H + sy) * W + sx] = (bf16)v;
        }
    }
}

// ---------------------------------------------------------------------------
// FC layer: out[p][o] = act[p][:] . w[o][:] + bias[o]  (optional ReLU)
// M = O (tile 64), N = batch (tile 32), K constexpr (512), chunks of 32.
// B staging is a pure bf16 copy -> async-to-LDS when the builtin exists.
// ---------------------------------------------------------------------------
template<bool RELU, typename TOUT, int K>
__global__ __launch_bounds__(256)
void fc_wmma_kernel(const bf16* __restrict__ act, const float* __restrict__ w,
                    const float* __restrict__ bias, TOUT* __restrict__ out, int O)
{
    __shared__ __align__(16) bf16 As[64 * 40];
    __shared__ __align__(16) bf16 Bs[32 * 40];

    const int t      = threadIdx.x;
    const int lane   = t & 31;
    const int wv     = t >> 5;
    const int waveOc = wv & 3;
    const int wavePx = wv >> 2;

    const int pBase  = blockIdx.x * 32;   // Nb = 256 -> 8 blocks, no guard
    const int ocBase = blockIdx.y * 64;

    const int pl = t >> 3;                // 0..31 batch row
    const int kb = (t & 7) * 4;           // 4 bf16 = 8 B per thread
    const int p  = pBase + pl;

    const int ar  = t >> 2;
    const int ac  = (t & 3) * 8;
    const int aoc = ocBase + ar;

    const int m = lane & 15, half = lane >> 4;
    const bf16* arow = &As[(waveOc * 16 + m) * 40];
    const bf16* brow = &Bs[(wavePx * 16 + m) * 40];

    v8f acc = {};

    for (int k0 = 0; k0 < K; k0 += 32) {
        __syncthreads();
        // ---- stage B: raw bf16 copy, 8 B per thread ----
#if USE_ASYNC
        {
            unsigned long long ga = (unsigned long long)(const void*)(act + (size_t)p * K + k0 + kb);
            unsigned           la = (unsigned)(unsigned long long)(const void*)&Bs[pl * 40 + kb];
            __builtin_amdgcn_global_load_async_to_lds_b64((gv2i*)ga, (lv2i*)(size_t)la, 0, 0);
        }
#else
        *(uint2*)&Bs[pl * 40 + kb] = *(const uint2*)(act + (size_t)p * K + k0 + kb);
#endif
        // ---- stage A: weights fp32 -> bf16 (row clamp keeps loads unpredicated) ----
        {
            const int row = (aoc < O) ? aoc : 0;
            const float4 f0 = *(const float4*)(w + row * K + k0 + ac);
            const float4 f1 = *(const float4*)(w + row * K + k0 + ac + 4);
            union { bf16 h[8]; uint4 q; } ua;
            ua.h[0] = (bf16)f0.x; ua.h[1] = (bf16)f0.y;
            ua.h[2] = (bf16)f0.z; ua.h[3] = (bf16)f0.w;
            ua.h[4] = (bf16)f1.x; ua.h[5] = (bf16)f1.y;
            ua.h[6] = (bf16)f1.z; ua.h[7] = (bf16)f1.w;
            if (aoc >= O) { ua.q.x = 0; ua.q.y = 0; ua.q.z = 0; ua.q.w = 0; }
            *(uint4*)&As[ar * 40 + ac] = ua.q;
        }
#if USE_ASYNC
        __builtin_amdgcn_s_wait_asynccnt(0);
#endif
        __syncthreads();

        v16bf a = ld_frag(arow + half * 8,  arow + 16 + half * 8);
        v16bf b = ld_frag(brow + half * 8, brow + 16 + half * 8);
        acc = __builtin_amdgcn_wmma_f32_16x16x32_bf16(false, a, false, b,
                                                      (short)0, acc, false, false);
    }

    const int sp = pBase + wavePx * 16 + m;
    #pragma unroll
    for (int r = 0; r < 8; ++r) {
        int oc  = ocBase + waveOc * 16 + ((lane < 16) ? r : r + 8);
        int occ = (oc < O) ? oc : 0;
        float v = acc[r] + bias[occ];
        if (RELU) v = v > 0.0f ? v : 0.0f;
        if (oc < O) out[sp * O + occ] = (TOUT)v;
    }
}

// ---------------------------------------------------------------------------
// 2x2 maxpool, stride 2, bf16 NCHW.
// ---------------------------------------------------------------------------
__global__ void maxpool2_kernel(const bf16* __restrict__ in, bf16* __restrict__ out,
                                int N, int C, int H)
{
    const int W = H;
    const int Ho = H >> 1, Wo = W >> 1;
    const int total = N * C * Ho * Wo;
    int i = blockIdx.x * blockDim.x + threadIdx.x;
    if (i >= total) return;
    int wo = i % Wo; int tmp = i / Wo;
    int ho = tmp % Ho; tmp /= Ho;
    int c  = tmp % C;  int n = tmp / C;
    const bf16* base = in + ((n * C + c) * H + ho * 2) * W + wo * 2;
    float a = (float)base[0], b = (float)base[1];
    float cc = (float)base[W], d = (float)base[W + 1];
    out[i] = (bf16)fmaxf(fmaxf(a, b), fmaxf(cc, d));
}

// ---------------------------------------------------------------------------
extern "C" void kernel_launch(void* const* d_in, const int* in_sizes, int n_in,
                              void* d_out, int out_size, void* d_ws, size_t ws_size,
                              hipStream_t stream)
{
    (void)in_sizes; (void)n_in; (void)out_size; (void)ws_size;

    const float* x = (const float*)d_in[0];
    const float* cw[13];
    for (int i = 0; i < 13; ++i) cw[i] = (const float*)d_in[1 + i];
    const float* fw[3];
    for (int i = 0; i < 3;  ++i) fw[i] = (const float*)d_in[14 + i];
    const float* fb[3];
    for (int i = 0; i < 3;  ++i) fb[i] = (const float*)d_in[17 + i];

    // ping-pong bf16 activation buffers; max activation = 256*64*32*32 elems
    const size_t ACT_ELEMS = (size_t)256 * 64 * 32 * 32;  // 16.78 M
    bf16* A = (bf16*)d_ws;
    bf16* B = A + ACT_ELEMS;

    const int NB = 256;
    #define CONV(TIN, Cc, Hh, SRC, WW, DST, OCC) \
        conv3x3_wmma_kernel<TIN, Cc, Hh><<<dim3(NB * Hh * Hh / 64, (OCC) / 64), 256, 0, stream>>>(SRC, WW, DST, OCC)
    #define POOL(SRC, DST, Cc, Hh) \
        maxpool2_kernel<<<(NB * (Cc) * ((Hh)/2) * ((Hh)/2) + 255) / 256, 256, 0, stream>>>(SRC, DST, NB, Cc, Hh)

    CONV(float, 3,   32, x, cw[0],  A, 64);
    CONV(bf16,  64,  32, A, cw[1],  B, 64);
    POOL(B, A, 64, 32);                         // -> 16x16
    CONV(bf16,  64,  16, A, cw[2],  B, 128);
    CONV(bf16,  128, 16, B, cw[3],  A, 128);
    POOL(A, B, 128, 16);                        // -> 8x8
    CONV(bf16,  128, 8,  B, cw[4],  A, 256);
    CONV(bf16,  256, 8,  A, cw[5],  B, 256);
    CONV(bf16,  256, 8,  B, cw[6],  A, 256);
    POOL(A, B, 256, 8);                         // -> 4x4
    CONV(bf16,  256, 4,  B, cw[7],  A, 512);
    CONV(bf16,  512, 4,  A, cw[8],  B, 512);
    CONV(bf16,  512, 4,  B, cw[9],  A, 512);
    POOL(A, B, 512, 4);                         // -> 2x2
    CONV(bf16,  512, 2,  B, cw[10], A, 512);
    CONV(bf16,  512, 2,  A, cw[11], B, 512);
    CONV(bf16,  512, 2,  B, cw[12], A, 512);
    POOL(A, B, 512, 2);                         // -> [256, 512] in B

    fc_wmma_kernel<true,  bf16,  512><<<dim3(8, 8), 256, 0, stream>>>(B, fw[0], fb[0], A, 512);
    fc_wmma_kernel<true,  bf16,  512><<<dim3(8, 8), 256, 0, stream>>>(A, fw[1], fb[1], B, 512);
    fc_wmma_kernel<false, float, 512><<<dim3(8, 1), 256, 0, stream>>>(B, fw[2], fb[2],
                                                                      (float*)d_out, 10);
    #undef CONV
    #undef POOL
}
